// deform_inconv_17686675325477
// MI455X (gfx1250) — compile-verified, matrix operand order checked by reading, not run on hardware
//
#include <hip/hip_runtime.h>
#include <hip/hip_bf16.h>

typedef _Float16 half8 __attribute__((ext_vector_type(8)));
typedef _Float16 v16h  __attribute__((ext_vector_type(16)));
typedef float    v8f   __attribute__((ext_vector_type(8)));

#define B_   4
#define C_   3
#define H_   512
#define W_   512
#define HW_  (H_ * W_)            // 262144 = 1<<18
#define OC_  64
#define NPIX (B_ * HW_)           // 1048576
#define STAT_BLOCKS 1024

// ---------------------------------------------------------------------------
// 1) Offsets conv: x(4,3,512,512) * Woff(6,3,3,3) -> offs(4,6,512,512), SAME pad
// ---------------------------------------------------------------------------
__global__ void k_offsets(const float* __restrict__ x, const float* __restrict__ wof,
                          float* __restrict__ offs) {
    int t = blockIdx.x * 256 + threadIdx.x;            // over B*6*HW
    int sp = t & (HW_ - 1);
    int oc = (t >> 18) % 6;
    int b  = t / (6 * HW_);
    int yy = sp >> 9, xx = sp & 511;
    float acc = 0.f;
    for (int ic = 0; ic < 3; ++ic)
        for (int ky = 0; ky < 3; ++ky) {
            int iy = yy + ky - 1;
            if (iy < 0 || iy >= H_) continue;
            for (int kx = 0; kx < 3; ++kx) {
                int ix = xx + kx - 1;
                if (ix < 0 || ix >= W_) continue;
                acc += x[((size_t)(b * 3 + ic) << 18) + iy * W_ + ix] *
                       wof[((oc * 3 + ic) * 3 + ky) * 3 + kx];
            }
        }
    offs[t] = acc;
}

// ---------------------------------------------------------------------------
// 2) Deformable bilinear gather (reference's raw .view(B,-1,2) reshape kept)
// ---------------------------------------------------------------------------
__global__ void k_deform(const float* __restrict__ x, const float* __restrict__ offs,
                         float* __restrict__ dfm) {
    int t = blockIdx.x * 256 + threadIdx.x;            // over B*C*HW
    int n  = t & (HW_ - 1);
    int bc = t >> 18;
    int b = bc / 3, c = bc % 3;
    const float* ob = offs + ((size_t)(b * 6) << 18);
    int m0 = 2 * n, m1 = 2 * n + 1;
    float yoff = ob[((size_t)(2 * c + (m0 >> 18)) << 18) + (m0 & (HW_ - 1))];
    float xoff = ob[((size_t)(2 * c + (m1 >> 18)) << 18) + (m1 & (HW_ - 1))];
    float fy = (float)(n >> 9)  + yoff;
    float fx = (float)(n & 511) + xoff;
    fy = fminf(fmaxf(fy, 0.f), (float)(H_ - 1));
    fx = fminf(fmaxf(fx, 0.f), (float)(W_ - 1));
    float y0 = floorf(fy), x0 = floorf(fx);
    float ay = fy - y0,   ax = fx - x0;
    int iy0 = (int)y0, ix0 = (int)x0;
    int iy1 = (int)ceilf(fy), ix1 = (int)ceilf(fx);
    const float* fb = x + ((size_t)bc << 18);
    float vlt = fb[iy0 * W_ + ix0];
    float vrt = fb[iy1 * W_ + ix0];
    float vlb = fb[iy0 * W_ + ix1];
    float vrb = fb[iy1 * W_ + ix1];
    float vt = vlt + (vrt - vlt) * ay;
    float vb = vlb + (vrb - vlb) * ay;
    dfm[t] = vt + (vb - vt) * ax;
}

// ---------------------------------------------------------------------------
// Fragment K-index mapping (A 16x32 f16 per ISA table):
//   lane<16  : elems 0..7 -> K 0..7,  elems 8..15 -> K 16..23
//   lane>=16 : elems 0..7 -> K 8..15, elems 8..15 -> K 24..31
// ---------------------------------------------------------------------------

// 3a) Pack W1 (64,3,3,3) into B fragments; K = ic*9 + ky*3 + kx, padded to 32.
__global__ void k_pack_w1(const float* __restrict__ W1, _Float16* __restrict__ wp) {
    int t = blockIdx.x * 256 + threadIdx.x;            // 4*32*16 = 2048
    int j    = t & 15;
    int lane = (t >> 4) & 31;
    int nc   = t >> 9;
    int hs = lane >> 4, nl = lane & 15;
    int k = (j < 8) ? (hs * 8 + j) : (16 + hs * 8 + (j - 8));
    float v = 0.f;
    if (k < 27) {
        int ic = k / 9, r = k - ic * 9;
        int ky = r / 3, kx = r - ky * 3;
        int o = nc * 16 + nl;
        v = W1[((o * 3 + ic) * 3 + ky) * 3 + kx];
    }
    wp[t] = (_Float16)v;
}

// 3b) Pack W2 (64,64,3,3) into B fragments, per (tap, kchunk, nchunk).
__global__ void k_pack_w2(const float* __restrict__ W2, _Float16* __restrict__ wp) {
    int t = blockIdx.x * 256 + threadIdx.x;            // 9*2*4*32*16 = 36864
    int j    = t & 15;
    int lane = (t >> 4) & 31;
    int nc   = (t >> 9) & 3;
    int kc   = (t >> 11) & 1;
    int tap  = t >> 12;
    int hs = lane >> 4, nl = lane & 15;
    int klocal = (j < 8) ? (hs * 8 + j) : (16 + hs * 8 + (j - 8));
    int ich = kc * 32 + klocal;
    int o   = nc * 16 + nl;
    int ky = tap / 3, kx = tap - ky * 3;
    wp[t] = (_Float16)W2[((o * 64 + ich) * 3 + ky) * 3 + kx];
}

// ---------------------------------------------------------------------------
// 4) conv1 3->64 via WMMA implicit GEMM. Wave tile: 16 pixels x 64 out-ch,
//    K = 27 (padded 32). Output NHWC f32.
// ---------------------------------------------------------------------------
__global__ void k_conv1_wmma(const float* __restrict__ dfm, const _Float16* __restrict__ wp,
                             const float* __restrict__ bias, float* __restrict__ out) {
    int lane = threadIdx.x & 31;
    int wv   = threadIdx.x >> 5;
    int gw   = blockIdx.x * 8 + wv;
    int P    = gw * 16;                                // tile pixel base (row-aligned)
    int b    = P >> 18;
    int sp   = P & (HW_ - 1);
    int y    = sp >> 9;
    int x0   = sp & 511;
    int nl = lane & 15, hs = lane >> 4;
    int px = x0 + nl;

    v16h a;
#pragma unroll
    for (int j = 0; j < 16; ++j) {
        int k = (j < 8) ? (hs * 8 + j) : (16 + hs * 8 + (j - 8));
        float v = 0.f;
        if (k < 27) {
            int ic = k / 9, r = k - ic * 9;
            int ky = r / 3, kx = r - ky * 3;
            int iy = y + ky - 1, ix = px + kx - 1;
            if (iy >= 0 && iy < H_ && ix >= 0 && ix < W_)
                v = dfm[((size_t)(b * 3 + ic) << 18) + iy * W_ + ix];
        }
        a[j] = (_Float16)v;
    }

    v8f acc[4];
#pragma unroll
    for (int nc = 0; nc < 4; ++nc) {
        float bv = bias[nc * 16 + nl];
#pragma unroll
        for (int r = 0; r < 8; ++r) acc[nc][r] = bv;
    }

#pragma unroll
    for (int nc = 0; nc < 4; ++nc) {
        v16h bf = *(const v16h*)(wp + ((nc * 32 + lane) << 4));
        acc[nc] = __builtin_amdgcn_wmma_f32_16x16x32_f16(
            false, a, false, bf, (short)0, acc[nc], false, false);
    }

#pragma unroll
    for (int nc = 0; nc < 4; ++nc) {
        int ch = nc * 16 + nl;
#pragma unroll
        for (int r = 0; r < 8; ++r) {
            int pl = r + hs * 8;                       // D row M = r + 8*(lane>=16)
            out[(((size_t)(P + pl)) << 6) + ch] = acc[nc][r];
        }
    }
}

// ---------------------------------------------------------------------------
// 5) Deterministic per-channel stats over NHWC f32: block partial sums.
// ---------------------------------------------------------------------------
__global__ void k_stats_nhwc(const float* __restrict__ buf, float* __restrict__ part) {
    __shared__ float ls[4][64], lq[4][64];
    int ch  = threadIdx.x & 63;
    int sub = threadIdx.x >> 6;
    float s = 0.f, q = 0.f;
    for (int p = blockIdx.x * 4 + sub; p < NPIX; p += STAT_BLOCKS * 4) {
        float v = buf[((size_t)p << 6) + ch];
        s += v; q += v * v;
    }
    ls[sub][ch] = s; lq[sub][ch] = q;
    __syncthreads();
    if (threadIdx.x < 64) {
        float S = ls[0][ch] + ls[1][ch] + ls[2][ch] + ls[3][ch];
        float Q = lq[0][ch] + lq[1][ch] + lq[2][ch] + lq[3][ch];
        part[blockIdx.x * 128 + ch]      = S;
        part[blockIdx.x * 128 + 64 + ch] = Q;
    }
}

__global__ void k_finalize(const float* __restrict__ part, const float* __restrict__ g,
                           const float* __restrict__ be, float* __restrict__ ss) {
    int ch = threadIdx.x;
    if (ch >= 64) return;
    float s = 0.f, q = 0.f;
    for (int i = 0; i < STAT_BLOCKS; ++i) {
        s += part[i * 128 + ch];
        q += part[i * 128 + 64 + ch];
    }
    const float inv_n = 1.0f / (float)NPIX;
    float mu  = s * inv_n;
    float var = q * inv_n - mu * mu;
    float rs  = rsqrtf(var + 1e-5f);
    float sc  = g[ch] * rs;
    ss[ch]      = sc;
    ss[64 + ch] = be[ch] - mu * sc;
}

// ---------------------------------------------------------------------------
// 6) BN+ReLU and convert to NHWC f16 for the WMMA conv2.
// ---------------------------------------------------------------------------
__global__ void k_bnrelu_f16(const float* __restrict__ in, const float* __restrict__ ss,
                             _Float16* __restrict__ out) {
    size_t t = (size_t)blockIdx.x * 256 + threadIdx.x; // NPIX*64
    int ch = (int)(t & 63);
    float v = fmaf(in[t], ss[ch], ss[64 + ch]);
    out[t] = (_Float16)fmaxf(v, 0.f);
}

// ---------------------------------------------------------------------------
// 7) conv2 64->64 via WMMA implicit GEMM. K = 9 taps x 64 in-ch (18 k-steps),
//    4 N-chunks -> 72 v_wmma per wave tile. Input NHWC f16, output NHWC f32.
// ---------------------------------------------------------------------------
__global__ void k_conv2_wmma(const _Float16* __restrict__ inh, const _Float16* __restrict__ wp,
                             const float* __restrict__ bias, float* __restrict__ out) {
    int lane = threadIdx.x & 31;
    int wv   = threadIdx.x >> 5;
    int gw   = blockIdx.x * 8 + wv;
    int P    = gw * 16;
    int b    = P >> 18;
    int sp   = P & (HW_ - 1);
    int y    = sp >> 9;
    int x0   = sp & 511;
    int nl = lane & 15, hs = lane >> 4;

    v8f acc[4];
#pragma unroll
    for (int nc = 0; nc < 4; ++nc) {
        float bv = bias[nc * 16 + nl];
#pragma unroll
        for (int r = 0; r < 8; ++r) acc[nc][r] = bv;
    }

#pragma unroll
    for (int ky = 0; ky < 3; ++ky) {
        int iy = y + ky - 1;
        if (iy < 0 || iy >= H_) continue;              // wave-uniform skip; EXEC stays full
#pragma unroll
        for (int kx = 0; kx < 3; ++kx) {
            int ix = x0 + nl + kx - 1;
            bool inb = (ix >= 0) && (ix < W_);
            const _Float16* bp = inh + (((size_t)(b * HW_ + iy * W_ + ix)) << 6);
            int tap = ky * 3 + kx;
#pragma unroll
            for (int kc = 0; kc < 2; ++kc) {
                half8 lo = {};
                half8 hi = {};
                if (inb) {                              // per-lane predicated b128 loads
                    lo = *(const half8*)(bp + kc * 32 + hs * 8);
                    hi = *(const half8*)(bp + kc * 32 + hs * 8 + 16);
                }
                v16h a;
#pragma unroll
                for (int j = 0; j < 8; ++j) { a[j] = lo[j]; a[j + 8] = hi[j]; }
#pragma unroll
                for (int nc = 0; nc < 4; ++nc) {
                    v16h bf = *(const v16h*)(wp + ((((tap * 2 + kc) * 4 + nc) * 32 + lane) << 4));
                    acc[nc] = __builtin_amdgcn_wmma_f32_16x16x32_f16(
                        false, a, false, bf, (short)0, acc[nc], false, false);
                }
            }
        }
    }

#pragma unroll
    for (int nc = 0; nc < 4; ++nc) {
        int ch = nc * 16 + nl;
#pragma unroll
        for (int r = 0; r < 8; ++r) {
            int pl = r + hs * 8;
            out[(((size_t)(P + pl)) << 6) + ch] = acc[nc][r];
        }
    }
}

// ---------------------------------------------------------------------------
// 8) BN+ReLU + NHWC->NCHW transpose through LDS (both sides coalesced).
// ---------------------------------------------------------------------------
__global__ void k_out_nchw(const float* __restrict__ in, const float* __restrict__ ss,
                           float* __restrict__ out) {
    __shared__ float tile[64 * 65];
    int tb = blockIdx.x;                               // 64-pixel tile
    int b  = tb >> 12;                                 // (tb*64)>>18
    int sp_base = (tb & 4095) * 64;
#pragma unroll
    for (int i = 0; i < 16; ++i) {
        int e = threadIdx.x + i * 256;
        int px = e >> 6, ch = e & 63;
        float v = in[((size_t)(tb * 64 + px) << 6) + ch];
        v = fmaf(v, ss[ch], ss[64 + ch]);
        tile[ch * 65 + px] = fmaxf(v, 0.f);
    }
    __syncthreads();
#pragma unroll
    for (int i = 0; i < 16; ++i) {
        int e = threadIdx.x + i * 256;
        int ch = e >> 6, px = e & 63;
        out[(((size_t)(b * 64 + ch)) << 18) + sp_base + px] = tile[ch * 65 + px];
    }
}

// ---------------------------------------------------------------------------
extern "C" void kernel_launch(void* const* d_in, const int* in_sizes, int n_in,
                              void* d_out, int out_size, void* d_ws, size_t ws_size,
                              hipStream_t stream) {
    (void)in_sizes; (void)n_in; (void)out_size; (void)ws_size;
    const float* x    = (const float*)d_in[0];
    const float* Woff = (const float*)d_in[1];
    const float* W1   = (const float*)d_in[2];
    const float* b1   = (const float*)d_in[3];
    const float* g1   = (const float*)d_in[4];
    const float* be1  = (const float*)d_in[5];
    const float* W2   = (const float*)d_in[6];
    const float* b2   = (const float*)d_in[7];
    const float* g2   = (const float*)d_in[8];
    const float* be2  = (const float*)d_in[9];
    float* out = (float*)d_out;
    char*  ws  = (char*)d_ws;

    const size_t R0 = (size_t)NPIX * 64 * sizeof(float);     // 268 MB: y1raw, then y2raw
    const size_t R1 = (size_t)NPIX * 64 * sizeof(_Float16);  // 134 MB: offs+dfm, then y1h

    float*    r0   = (float*)ws;                             // NHWC f32 conv outputs
    float*    offs = (float*)(ws + R0);                      // 25 MB
    float*    dfm  = (float*)(ws + R0 + (size_t)B_ * 6 * HW_ * sizeof(float)); // 12.6 MB
    _Float16* y1h  = (_Float16*)(ws + R0);                   // reuses offs/dfm region
    char*     tail = ws + R0 + R1;
    _Float16* w1p  = (_Float16*)tail;                        // 4 KB
    _Float16* w2p  = (_Float16*)(tail + 4096);               // 72 KB
    float*    ss1  = (float*)(tail + 4096 + 73728);          // scale/shift layer 1
    float*    ss2  = ss1 + 128;
    float*    part = ss2 + 128;                              // 512 KB stat partials

    k_pack_w1<<<2048 / 256, 256, 0, stream>>>(W1, w1p);
    k_pack_w2<<<36864 / 256, 256, 0, stream>>>(W2, w2p);
    k_offsets<<<(B_ * 6 * HW_) / 256, 256, 0, stream>>>(x, Woff, offs);
    k_deform<<<(B_ * C_ * HW_) / 256, 256, 0, stream>>>(x, offs, dfm);

    k_conv1_wmma<<<NPIX / 128, 256, 0, stream>>>(dfm, w1p, b1, r0);
    k_stats_nhwc<<<STAT_BLOCKS, 256, 0, stream>>>(r0, part);
    k_finalize<<<1, 64, 0, stream>>>(part, g1, be1, ss1);
    k_bnrelu_f16<<<((size_t)NPIX * 64) / 256, 256, 0, stream>>>(r0, ss1, y1h);

    k_conv2_wmma<<<NPIX / 128, 256, 0, stream>>>(y1h, w2p, b2, r0);
    k_stats_nhwc<<<STAT_BLOCKS, 256, 0, stream>>>(r0, part);
    k_finalize<<<1, 64, 0, stream>>>(part, g2, be2, ss2);
    k_out_nchw<<<NPIX / 64, 256, 0, stream>>>(r0, ss2, out);
}
